// SpatialBayesianLayer_21706764714427
// MI455X (gfx1250) — 8-bit matrix operand order repaired and checked by reading, compiled, not run on hardware
//
#include <hip/hip_runtime.h>
#include <stdint.h>


typedef __attribute__((ext_vector_type(8))) int v8i;

#define QBF   127.0f
#define EPSF  1e-5f
#define FDIM  128
#define SDIM  32768
#define BDIM  8
#define NTOK  (BDIM * SDIM)
#define WAVES 4
#define MAGIC 12582912.0f


__device__ __forceinline__ uint32_t quant4(const float4 v, const float s) {
#pragma clang fp contract(off)
  const float m0 = (v.x * s) + MAGIC;
  const float m1 = (v.y * s) + MAGIC;
  const float m2 = (v.z * s) + MAGIC;
  const float m3 = (v.w * s) + MAGIC;
  const uint32_t b01 = __builtin_amdgcn_perm(
      __float_as_uint(m1), __float_as_uint(m0), 0x00000400u);
  const uint32_t b23 = __builtin_amdgcn_perm(
      __float_as_uint(m3), __float_as_uint(m2), 0x00000400u);
  return __builtin_amdgcn_perm(b23, b01, 0x05040100u);
}

__global__ __launch_bounds__(256) void prep_kernel(
    const float* __restrict__ W1, const float* __restrict__ W2,
    float* __restrict__ gsums, float* __restrict__ wdq,
    uint32_t* __restrict__ w1p, uint32_t* __restrict__ w2p)
{
  __shared__ float red[256];
  const int tid = threadIdx.x;

  for (int i = tid; i < BDIM * FDIM; i += 256) gsums[i] = 0.f;

  float part = 0.f;
  for (int i = tid; i < FDIM * 2 * FDIM; i += 256) part += fabsf(W1[i]);
  red[tid] = part;
  __syncthreads();
  for (int s = 128; s > 0; s >>= 1) {
    if (tid < s) red[tid] += red[tid + s];
    __syncthreads();
  }
  const float w1dq = fmaxf(red[0] * (1.f / 32768.f), EPSF);
  const float ws1  = 1.f / w1dq;
  __syncthreads();

  part = 0.f;
  for (int i = tid; i < FDIM * FDIM; i += 256) part += fabsf(W2[i]);
  red[tid] = part;
  __syncthreads();
  for (int s = 128; s > 0; s >>= 1) {
    if (tid < s) red[tid] += red[tid + s];
    __syncthreads();
  }
  const float w2dq = fmaxf(red[0] * (1.f / 16384.f), EPSF);
  const float ws2  = 1.f / w2dq;

  if (tid == 0) { wdq[0] = w1dq; wdq[1] = w2dq; }

  for (int idx = tid; idx < 8192; idx += 256) {
    const int p = idx & 7, l = (idx >> 3) & 31, c = (idx >> 8) & 3, t = idx >> 10;
    const int n = l & 15, hi = l >> 4;
    uint32_t u = 0;
    #pragma unroll
    for (int b = 0; b < 4; ++b) {
      const int k = c * 64 + (p >> 1) * 16 + hi * 8 + (p & 1) * 4 + b;
      const float w = W1[(t * 16 + n) * 256 + k];
      int q = (int)rintf(w * ws1);
      q = q < -1 ? -1 : (q > 1 ? 1 : q);
      u |= ((uint32_t)(q & 0xFF)) << (8 * b);
    }
    w1p[idx] = u;
  }
  for (int idx = tid; idx < 4096; idx += 256) {
    const int p = idx & 7, l = (idx >> 3) & 31, c = (idx >> 8) & 1, t = idx >> 9;
    const int n = l & 15, hi = l >> 4;
    uint32_t u = 0;
    #pragma unroll
    for (int b = 0; b < 4; ++b) {
      const int k = c * 64 + (p >> 1) * 16 + hi * 8 + (p & 1) * 4 + b;
      const float w = W2[(t * 16 + n) * 128 + k];
      int q = (int)rintf(w * ws2);
      q = q < -1 ? -1 : (q > 1 ? 1 : q);
      u |= ((uint32_t)(q & 0xFF)) << (8 * b);
    }
    w2p[idx] = u;
  }
}

__global__ __launch_bounds__(WAVES * 32) void bitnet_fused(
    const float* __restrict__ evidence, const float* __restrict__ prior,
    const float* __restrict__ b1, const float* __restrict__ b2,
    const float* __restrict__ wdq,
    const uint32_t* __restrict__ w1p, const uint32_t* __restrict__ w2p,
    float* __restrict__ gsums, float* __restrict__ out)
{
  __shared__ float hbuf[WAVES][16][FDIM];
  __shared__ float pbuf[WAVES][16][FDIM];
  __shared__ float lsum[FDIM];

  const int tid  = threadIdx.x;
  const int wave = tid >> 5;
  const int lane = tid & 31;
  const int row  = lane & 15;
  const int hi   = lane >> 4;

  if (tid < FDIM) lsum[tid] = 0.f;
  __syncthreads();

  const int tile_tok = (blockIdx.x * WAVES + wave) * 16;
  const size_t mytok = (size_t)(tile_tok + row);
  const float w1dq = wdq[0];
  const float w2dq = wdq[1];

  float amax = 0.f;
  #pragma unroll
  for (int c = 0; c < 4; ++c) {
    #pragma unroll
    for (int p = 0; p < 8; ++p) {
      const int k = c * 64 + ((p >> 1) << 4) + (hi << 3) + ((p & 1) << 2);
      const float* src = (k < FDIM) ? prior : evidence;
      const float4 v = *(const float4*)(src + mytok * FDIM + (k & (FDIM - 1)));
      amax = fmaxf(amax, fmaxf(fmaxf(fabsf(v.x), fabsf(v.y)),
                               fmaxf(fabsf(v.z), fabsf(v.w))));
    }
  }
  amax = fmaxf(amax, __shfl_xor(amax, 16, 32));
  const float sA    = QBF / fmaxf(amax, EPSF);
  const float invsA = 1.f / sA;

  v8i a1[4];
  #pragma unroll
  for (int c = 0; c < 4; ++c) {
    #pragma unroll
    for (int p = 0; p < 8; ++p) {
      const int k = c * 64 + ((p >> 1) << 4) + (hi << 3) + ((p & 1) << 2);
      const float* src = (k < FDIM) ? prior : evidence;
      const float4 v = *(const float4*)(src + mytok * FDIM + (k & (FDIM - 1)));
      if (k < FDIM)
        *(float4*)&pbuf[wave][row][k] = v;
      a1[c][p] = (int)quant4(v, sA);
    }
  }

  float g1[8];
  #pragma unroll
  for (int r = 0; r < 8; ++r) g1[r] = __shfl(invsA, hi * 8 + r, 32) * w1dq;

  #pragma unroll
  for (int t = 0; t < 8; ++t) {
    v8i acc = {0, 0, 0, 0, 0, 0, 0, 0};
    #pragma unroll
    for (int c = 0; c < 4; ++c) {
      const v8i bv = *(const v8i*)(w1p + ((size_t)(t * 4 + c) * 32 + lane) * 8);
      acc = __builtin_amdgcn_wmma_i32_16x16x64_iu8(
          true, a1[c], true, bv, acc, false, false);
    }
    const int o = t * 16 + row;
    const float bias = b1[o];
    #pragma unroll
    for (int r = 0; r < 8; ++r) {
      const float h = (float)acc[r] * g1[r] + bias;
      hbuf[wave][hi * 8 + r][o] = fmaxf(h, 0.f);
    }
  }

  float amax2 = 0.f;
  #pragma unroll
  for (int c = 0; c < 2; ++c) {
    #pragma unroll
    for (int p = 0; p < 8; ++p) {
      const int k = c * 64 + ((p >> 1) << 4) + (hi << 3) + ((p & 1) << 2);
      const float4 v = *(const float4*)&hbuf[wave][row][k];
      amax2 = fmaxf(amax2, fmaxf(fmaxf(v.x, v.y), fmaxf(v.z, v.w)));
    }
  }
  amax2 = fmaxf(amax2, __shfl_xor(amax2, 16, 32));
  const float sB    = QBF / fmaxf(amax2, EPSF);
  const float invsB = 1.f / sB;

  v8i a2[2];
  #pragma unroll
  for (int c = 0; c < 2; ++c) {
    #pragma unroll
    for (int p = 0; p < 8; ++p) {
      const int k = c * 64 + ((p >> 1) << 4) + (hi << 3) + ((p & 1) << 2);
      const float4 v = *(const float4*)&hbuf[wave][row][k];
      a2[c][p] = (int)quant4(v, sB);
    }
  }
  float g2[8];
  #pragma unroll
  for (int r = 0; r < 8; ++r) g2[r] = __shfl(invsB, hi * 8 + r, 32) * w2dq;

  #pragma unroll
  for (int t = 0; t < 8; ++t) {
    v8i acc = {0, 0, 0, 0, 0, 0, 0, 0};
    #pragma unroll
    for (int c = 0; c < 2; ++c) {
      const v8i bv = *(const v8i*)(w2p + ((size_t)(t * 2 + c) * 32 + lane) * 8);
      acc = __builtin_amdgcn_wmma_i32_16x16x64_iu8(
          true, a2[c], true, bv, acc, false, false);
    }
    const int o = t * 16 + row;
    const float bias = b2[o];
    float colsum = 0.f;
    #pragma unroll
    for (int r = 0; r < 8; ++r) {
      const size_t tok = (size_t)(tile_tok + hi * 8 + r);
      const float logit = (float)acc[r] * g2[r] + bias;
      const float like  = __builtin_amdgcn_rcpf(1.f + __expf(-logit));
      const float pu    = pbuf[wave][hi * 8 + r][o] * like;
      out[tok * FDIM + o] = pu;
      colsum += pu;
    }
    colsum += __shfl_xor(colsum, 16, 32);
    if (hi == 0) atomicAdd(&lsum[o], colsum);
  }

  __syncthreads();
  if (tid < FDIM) {
    const int b = (blockIdx.x * (WAVES * 16)) >> 15;
    atomicAdd(&gsums[b * FDIM + tid], lsum[tid]);
  }
}

__global__ __launch_bounds__(256) void normalize_kernel(
    float* __restrict__ out, const float* __restrict__ gsums, int n4)
{
  const int stride = gridDim.x * blockDim.x;
  float4* o4 = (float4*)out;
  for (int i = blockIdx.x * blockDim.x + threadIdx.x; i < n4; i += stride) {
    const size_t e = (size_t)i * 4;
    const int b = (int)(e >> 22);
    const int f = (int)(e & (FDIM - 1));
    const float4 s4 = *(const float4*)(gsums + b * FDIM + f);
    float4 v = o4[i];
    v.x /= fmaxf(s4.x, 1e-10f);
    v.y /= fmaxf(s4.y, 1e-10f);
    v.z /= fmaxf(s4.z, 1e-10f);
    v.w /= fmaxf(s4.w, 1e-10f);
    o4[i] = v;
  }
}

extern "C" void kernel_launch(void* const* d_in, const int* in_sizes, int n_in,
                              void* d_out, int out_size, void* d_ws, size_t ws_size,
                              hipStream_t stream) {
  const float* evidence = (const float*)d_in[0];
  const float* prior    = (const float*)d_in[1];
  const float* W1       = (const float*)d_in[2];
  const float* b1       = (const float*)d_in[3];
  const float* W2       = (const float*)d_in[4];
  const float* b2       = (const float*)d_in[5];
  float* out = (float*)d_out;

  char* ws = (char*)d_ws;
  float*    gsums = (float*)ws;
  float*    wdq   = (float*)(ws + 4096);
  uint32_t* w1p   = (uint32_t*)(ws + 4352);
  uint32_t* w2p   = (uint32_t*)(ws + 4352 + 32768);

  prep_kernel<<<1, 256, 0, stream>>>(W1, W2, gsums, wdq, w1p, w2p);

  const int nblocks = NTOK / (WAVES * 16);
  bitnet_fused<<<nblocks, WAVES * 32, 0, stream>>>(
      evidence, prior, b1, b2, wdq, w1p, w2p, gsums, out);

  normalize_kernel<<<8192, 256, 0, stream>>>(out, gsums, out_size / 4);
}
